// Transformer_34033320853585
// MI455X (gfx1250) — compile-verified, matrix operand order checked by reading, not run on hardware
//
#include <hip/hip_runtime.h>
#include <hip/hip_bf16.h>

// ---------------- model dimensions ----------------
#define N_LAYER 4
#define D_MODEL 512
#define N_HEAD 8
#define D_HEAD 64
#define D_INNER 2048
#define KERN 9
#define POOL 8
#define N_CLASSES 10
#define BATCH 4
#define RAW_LEN 8192
#define S_LEN 1023              // (8192-9+1)/8
#define SB (S_LEN*BATCH)        // 4092 rows
#define R_LEN (2*S_LEN-1)       // 2045
#define BN (BATCH*N_HEAD)       // 32 batched attention GEMMs
#define CHUNK 128               // query-row chunk for attention scratch
#define NCHUNK 8                // ceil(1023/128)
#define LD_AC 1024
#define LD_BD 2048

typedef __attribute__((ext_vector_type(16))) _Float16 v16h;
typedef __attribute__((ext_vector_type(8)))  _Float16 v8h;
typedef __attribute__((ext_vector_type(8)))  float    v8f;
typedef __attribute__((ext_vector_type(4)))  float    f32x4;

// ---------------------------------------------------------------------------
// Branch-free panel stager for a 128x32 A panel and 32x128 B panel in WMMA
// fragment order (lane L -> M=L%16; element e -> K=(e/8)*16+(L/16)*8+e%8).
// Slot p = t covers fragment f = t/32, lane L = t%32, both 8-element halves.
// Loads are unconditional from clamped addresses (proven in-allocation) and
// masked with selects; K-masking only runs on a partial last K-step.
// ---------------------------------------------------------------------------
__device__ __forceinline__ void stage_panel(
    _Float16* __restrict__ Ald, _Float16* __restrict__ Bld,
    const float* __restrict__ A, const float* __restrict__ B,
    int M, int N, int K, int lda, int ldb, bool bT,
    int row0, int col0, int k0,
    int t, int frag, int lane16, int khalf)
{
  const bool kfull = (k0 + 32) <= K;
  const int grA = row0 + frag * 16 + lane16;
  const bool okA = grA < M;
  const float* arow = A + (long long)(okA ? grA : (M - 1)) * lda;
  const int gc  = col0 + frag * 16 + lane16;
  const bool okB = gc < N;
  const int cc  = okB ? gc : (N - 1);

#pragma unroll
  for (int h = 0; h < 2; ++h) {
    const int kbase = k0 + h * 16 + khalf;
    // ---- A: 8 K-contiguous floats ----
    {
      const float* src = arow + kbase;
      const f32x4 lo = *(const f32x4*)src;
      const f32x4 hi = *(const f32x4*)(src + 4);
      v8h o;
      if (kfull) {
#pragma unroll
        for (int j = 0; j < 4; ++j) {
          o[j]     = okA ? (_Float16)lo[j] : (_Float16)0.f;
          o[4 + j] = okA ? (_Float16)hi[j] : (_Float16)0.f;
        }
      } else {
#pragma unroll
        for (int j = 0; j < 4; ++j) {
          o[j]     = (okA && (kbase + j)     < K) ? (_Float16)lo[j] : (_Float16)0.f;
          o[4 + j] = (okA && (kbase + 4 + j) < K) ? (_Float16)hi[j] : (_Float16)0.f;
        }
      }
      *(v8h*)&Ald[(t << 4) + h * 8] = o;
    }
    // ---- B ----
    {
      v8h o;
      if (bT) {        // B[col][k]: K-contiguous
        const float* src = B + (long long)cc * ldb + kbase;
        const f32x4 lo = *(const f32x4*)src;
        const f32x4 hi = *(const f32x4*)(src + 4);
        if (kfull) {
#pragma unroll
          for (int j = 0; j < 4; ++j) {
            o[j]     = okB ? (_Float16)lo[j] : (_Float16)0.f;
            o[4 + j] = okB ? (_Float16)hi[j] : (_Float16)0.f;
          }
        } else {
#pragma unroll
          for (int j = 0; j < 4; ++j) {
            o[j]     = (okB && (kbase + j)     < K) ? (_Float16)lo[j] : (_Float16)0.f;
            o[4 + j] = (okB && (kbase + 4 + j) < K) ? (_Float16)hi[j] : (_Float16)0.f;
          }
        }
      } else {         // B[k][col]: 8 ldb-strided loads, k clamped
#pragma unroll
        for (int j = 0; j < 8; ++j) {
          const int kk = kbase + j;
          const int kc = (kk < K) ? kk : (K - 1);
          const float v = B[(long long)kc * ldb + cc];
          o[j] = (okB && kk < K) ? (_Float16)v : (_Float16)0.f;
        }
      }
      *(v8h*)&Bld[(t << 4) + h * 8] = o;
    }
  }
}

// ---------------------------------------------------------------------------
// Generic f32 GEMM via v_wmma_f32_16x16x32_f16, double-buffered LDS.
//   C[z] = op( A[z] (MxK) * B[z%bzMod] (KxN) ), f16 operands, f32 accumulate.
// Block: 256 threads = 8 wave32, 128x128 C tile, 8 WMMAs/wave/K-step.
// flags: bit0 = ReLU, bit1 = B transposed (read B[col*ldb+k]), bit2 = add Cin
// ---------------------------------------------------------------------------
__global__ __launch_bounds__(256) void gemm_wmma_kernel(
    const float* __restrict__ A, const float* __restrict__ B,
    const float* __restrict__ Cin, float* __restrict__ Cout,
    const float* __restrict__ bias,
    int M, int N, int K, int lda, int ldb, int ldc,
    long long sA, long long sB, long long sC,
    int bzMod, int flags)
{
  __shared__ __align__(32) _Float16 Al[2][128 * 32];
  __shared__ __align__(32) _Float16 Bl[2][128 * 32];

  const int t  = threadIdx.x;
  const int z  = blockIdx.z;
  const int zb = bzMod ? (z % bzMod) : z;
  A += (long long)z  * sA;
  B += (long long)zb * sB;
  const long long coff = (long long)z * sC;

  const int row0 = blockIdx.y * 128;
  const int col0 = blockIdx.x * 128;
  const bool bT  = (flags & 2) != 0;

  const int lane   = t & 31;
  const int frag   = t >> 5;            // loader fragment / wave id
  const int lane16 = lane & 15;
  const int khalf  = (lane >> 4) * 8;

  // wave compute mapping: 2 A-fragments x 4 B-fragments = 8 subtiles
  const int fm0 = (frag & 3) * 2;
  const int fn0 = (frag >> 2) * 4;

  v8f acc[8];
#pragma unroll
  for (int i = 0; i < 8; ++i) acc[i] = (v8f){0.f,0.f,0.f,0.f,0.f,0.f,0.f,0.f};

  stage_panel(Al[0], Bl[0], A, B, M, N, K, lda, ldb, bT,
              row0, col0, 0, t, frag, lane16, khalf);

  int buf = 0;
  for (int k0 = 0; k0 < K; k0 += 32) {
    __syncthreads();
    if (k0 + 32 < K)
      stage_panel(Al[buf ^ 1], Bl[buf ^ 1], A, B, M, N, K, lda, ldb, bT,
                  row0, col0, k0 + 32, t, frag, lane16, khalf);

    const v16h a0 = *(const v16h*)&Al[buf][((fm0    ) * 32 + lane) * 16];
    const v16h a1 = *(const v16h*)&Al[buf][((fm0 + 1) * 32 + lane) * 16];
    const v16h b0 = *(const v16h*)&Bl[buf][((fn0    ) * 32 + lane) * 16];
    const v16h b1 = *(const v16h*)&Bl[buf][((fn0 + 1) * 32 + lane) * 16];
    const v16h b2 = *(const v16h*)&Bl[buf][((fn0 + 2) * 32 + lane) * 16];
    const v16h b3 = *(const v16h*)&Bl[buf][((fn0 + 3) * 32 + lane) * 16];

    acc[0] = __builtin_amdgcn_wmma_f32_16x16x32_f16(false, a0, false, b0, (short)0, acc[0], false, false);
    acc[1] = __builtin_amdgcn_wmma_f32_16x16x32_f16(false, a0, false, b1, (short)0, acc[1], false, false);
    acc[2] = __builtin_amdgcn_wmma_f32_16x16x32_f16(false, a0, false, b2, (short)0, acc[2], false, false);
    acc[3] = __builtin_amdgcn_wmma_f32_16x16x32_f16(false, a0, false, b3, (short)0, acc[3], false, false);
    acc[4] = __builtin_amdgcn_wmma_f32_16x16x32_f16(false, a1, false, b0, (short)0, acc[4], false, false);
    acc[5] = __builtin_amdgcn_wmma_f32_16x16x32_f16(false, a1, false, b1, (short)0, acc[5], false, false);
    acc[6] = __builtin_amdgcn_wmma_f32_16x16x32_f16(false, a1, false, b2, (short)0, acc[6], false, false);
    acc[7] = __builtin_amdgcn_wmma_f32_16x16x32_f16(false, a1, false, b3, (short)0, acc[7], false, false);
    buf ^= 1;
  }

  // epilogue: lane L, vgpr r -> M = r + 8*(L/16), N = L%16
  const bool doBias = bias != nullptr;
  const bool doAdd  = (flags & 4) != 0;
  const bool doRelu = (flags & 1) != 0;
#pragma unroll
  for (int i = 0; i < 2; ++i) {
    const int rbase = row0 + (fm0 + i) * 16 + ((lane >> 4) << 3);
#pragma unroll
    for (int j = 0; j < 4; ++j) {
      const int col = col0 + (fn0 + j) * 16 + lane16;
      if (col < N) {
        const float bcol = doBias ? bias[col] : 0.f;
#pragma unroll
        for (int r = 0; r < 8; ++r) {
          const int row = rbase + r;
          if (row < M) {
            float v = acc[i * 4 + j][r] + bcol;
            if (doAdd)  v += Cin[coff + (long long)row * ldc + col];
            if (doRelu) v = fmaxf(v, 0.f);
            Cout[coff + (long long)row * ldc + col] = v;
          }
        }
      }
    }
  }
}

// ---------------------------------------------------------------------------
// conv1d(9) + ReLU + mean-pool(8) -> core[s][b][d]
// ---------------------------------------------------------------------------
__global__ __launch_bounds__(256) void conv_pool_kernel(
    const float* __restrict__ sig, const float* __restrict__ cw,
    const float* __restrict__ cb, float* __restrict__ core)
{
  const long long idx = (long long)blockIdx.x * 256 + threadIdx.x;
  if (idx >= (long long)S_LEN * BATCH * D_MODEL) return;
  const int d   = (int)(idx & (D_MODEL - 1));
  const int rem = (int)(idx >> 9);
  const int b   = rem & (BATCH - 1);
  const int s   = rem >> 2;
  const float bias = cb[d];
  const float* sp = sig + (long long)b * RAW_LEN + s * POOL;
  float acc = 0.f;
#pragma unroll
  for (int pI = 0; pI < POOL; ++pI) {
    float c = bias;
#pragma unroll
    for (int k = 0; k < KERN; ++k) c += sp[pI + k] * cw[k * D_MODEL + d];
    acc += fmaxf(c, 0.f);
  }
  core[idx] = acc * (1.f / POOL);
}

// ---------------------------------------------------------------------------
// relative position embedding: pe[j][d], j=0..2S-2, pos = (S-1)-j
// ---------------------------------------------------------------------------
__global__ __launch_bounds__(256) void posemb_kernel(float* __restrict__ pe)
{
  const int idx = blockIdx.x * 256 + threadIdx.x;
  if (idx >= R_LEN * D_MODEL) return;
  const int d = idx & (D_MODEL - 1);
  const int j = idx >> 9;
  const float pos = (float)(S_LEN - 1 - j);
  const int i2 = (d < 256) ? d : d - 256;
  const float invf = expf(-((float)i2 / 256.f) * 9.210340371976184f); // ln(1e4)
  const float ang = pos * invf;
  pe[idx] = (d < 256) ? sinf(ang) : cosf(ang);
}

// ---------------------------------------------------------------------------
// split heads -> qw/qr/kk/vv in [b][n][s][d] layout, with rel biases folded
// ---------------------------------------------------------------------------
__global__ __launch_bounds__(256) void split_qkv_kernel(
    const float* __restrict__ heads,
    const float* __restrict__ rwb, const float* __restrict__ rrb,
    float* __restrict__ qw, float* __restrict__ qr,
    float* __restrict__ kk, float* __restrict__ vv)
{
  const long long idx = (long long)blockIdx.x * 256 + threadIdx.x;
  if (idx >= (long long)BN * S_LEN * D_HEAD) return;
  const int d = (int)(idx & (D_HEAD - 1));
  const int s = (int)((idx >> 6) % S_LEN);
  const int z = (int)(idx / ((long long)S_LEN * D_HEAD));
  const int n = z & (N_HEAD - 1);
  const int b = z >> 3;
  const long long hrow = ((long long)s * BATCH + b) * (3 * N_HEAD * D_HEAD);
  const int hd = n * D_HEAD + d;
  const float q = heads[hrow + hd];
  const float k = heads[hrow + N_HEAD * D_HEAD + hd];
  const float v = heads[hrow + 2 * N_HEAD * D_HEAD + hd];
  qw[idx] = q + rwb[hd];
  qr[idx] = q + rrb[hd];
  kk[idx] = k;
  vv[idx] = v;
}

// ---------------------------------------------------------------------------
// fused rel-shift + scale + softmax over key axis; P overwrites AC rows.
// score[i][j] = (AC[i][j] + BD[i][S-1+j-i]) * scale
// ---------------------------------------------------------------------------
__global__ __launch_bounds__(256) void softmax_rel_kernel(
    float* __restrict__ AC, const float* __restrict__ BD, int i0)
{
  const int r = blockIdx.x;       // row within chunk
  const int z = blockIdx.y;       // (b,n) batch
  const int i = i0 + r;
  float* acrow = AC + ((long long)z * CHUNK + r) * LD_AC;
  const float* bdrow = BD + ((long long)z * CHUNK + r) * LD_BD + (S_LEN - 1 - i);
  const int t = threadIdx.x;
  __shared__ float red[256];
  const float scale = 0.125f; // 1/sqrt(64)

  float lmax = -3.0e38f;
  for (int j = t; j < S_LEN; j += 256) {
    const float v = (acrow[j] + bdrow[j]) * scale;
    acrow[j] = v;
    lmax = fmaxf(lmax, v);
  }
  red[t] = lmax; __syncthreads();
  for (int off = 128; off > 0; off >>= 1) {
    if (t < off) red[t] = fmaxf(red[t], red[t + off]);
    __syncthreads();
  }
  const float m = red[0]; __syncthreads();

  float lsum = 0.f;
  for (int j = t; j < S_LEN; j += 256) {
    const float e = __expf(acrow[j] - m);
    acrow[j] = e;
    lsum += e;
  }
  red[t] = lsum; __syncthreads();
  for (int off = 128; off > 0; off >>= 1) {
    if (t < off) red[t] += red[t + off];
    __syncthreads();
  }
  const float inv = 1.f / red[0];
  for (int j = t; j < S_LEN; j += 256) acrow[j] *= inv;
}

// ---------------------------------------------------------------------------
// [b][n][s][d] -> [(s*B+b)][n*64+d]
// ---------------------------------------------------------------------------
__global__ __launch_bounds__(256) void head_merge_kernel(
    const float* __restrict__ avec, float* __restrict__ avec2)
{
  const long long idx = (long long)blockIdx.x * 256 + threadIdx.x;
  if (idx >= (long long)SB * D_MODEL) return;
  const int d2 = (int)(idx & (D_MODEL - 1));
  const int i  = (int)(idx >> 9);
  const int b  = i & (BATCH - 1);
  const int s  = i >> 2;
  const int n  = d2 >> 6;
  const int d  = d2 & (D_HEAD - 1);
  avec2[idx] = avec[(((long long)(b * N_HEAD + n) * S_LEN + s) << 6) + d];
}

// ---------------------------------------------------------------------------
// classifier head: out[b][c] = core[S-1][b] . fc_w[:,c] + fc_b[c]
// ---------------------------------------------------------------------------
__global__ void fc_kernel(const float* __restrict__ core,
                          const float* __restrict__ fcw,
                          const float* __restrict__ fcb,
                          float* __restrict__ out)
{
  const int c = threadIdx.x;
  const int b = blockIdx.x;
  if (c >= N_CLASSES) return;
  const float* row = core + ((long long)(S_LEN - 1) * BATCH + b) * D_MODEL;
  float acc = fcb[c];
  for (int d = 0; d < D_MODEL; ++d) acc += row[d] * fcw[d * N_CLASSES + c];
  out[b * N_CLASSES + c] = acc;
}

// ---------------------------------------------------------------------------
static inline void gemm(hipStream_t st, const float* A, const float* B,
                        const float* Cin, float* Cout, const float* bias,
                        int M, int N, int K, int lda, int ldb, int ldc,
                        long long sA, long long sB, long long sC,
                        int nbatch, int bzMod, int flags)
{
  dim3 g((N + 127) / 128, (M + 127) / 128, nbatch);
  gemm_wmma_kernel<<<g, 256, 0, st>>>(A, B, Cin, Cout, bias, M, N, K,
                                      lda, ldb, ldc, sA, sB, sC, bzMod, flags);
}

extern "C" void kernel_launch(void* const* d_in, const int* in_sizes, int n_in,
                              void* d_out, int out_size, void* d_ws, size_t ws_size,
                              hipStream_t stream) {
  const float* signal = (const float*)d_in[0];
  const float* conv_w = (const float*)d_in[1];
  const float* conv_b = (const float*)d_in[2];
  const float* qkv_w  = (const float*)d_in[3];
  const float* r_w    = (const float*)d_in[4];
  const float* o_w    = (const float*)d_in[5];
  const float* ff1_w  = (const float*)d_in[6];
  const float* ff1_b  = (const float*)d_in[7];
  const float* ff2_w  = (const float*)d_in[8];
  const float* ff2_b  = (const float*)d_in[9];
  const float* r_w_bias = (const float*)d_in[10];
  const float* r_r_bias = (const float*)d_in[11];
  const float* fc_w   = (const float*)d_in[12];
  const float* fc_b   = (const float*)d_in[13];
  float* out = (float*)d_out;

  // ---- workspace layout (floats) ----
  float* ws = (float*)d_ws;
  const long long SZ_CORE  = (long long)SB * D_MODEL;          // 2,095,104
  const long long SZ_PE    = (long long)R_LEN * D_MODEL;       // 1,047,040
  const long long SZ_HEADS = (long long)SB * 3 * D_MODEL;      // 6,285,312
  const long long SZ_RK    = (long long)R_LEN * D_MODEL;
  const long long SZ_HSD   = (long long)BN * S_LEN * D_HEAD;   // 2,095,104
  const long long SZ_AC    = (long long)BN * CHUNK * LD_AC;    // 4,194,304
  const long long SZ_BD    = (long long)BN * CHUNK * LD_BD;    // 8,388,608
  const long long SZ_FFH   = (long long)SB * D_INNER;          // 8,380,416

  long long off = 0;
  float* core  = ws + off; off += SZ_CORE;
  float* pe    = ws + off; off += SZ_PE;
  float* heads = ws + off; off += SZ_HEADS;
  float* rk    = ws + off; off += SZ_RK;
  float* qw    = ws + off; off += SZ_HSD;
  float* qr    = ws + off; off += SZ_HSD;
  float* kk    = ws + off; off += SZ_HSD;
  float* vv    = ws + off; off += SZ_HSD;
  float* acb   = ws + off; off += SZ_AC;
  float* bdb   = ws + off; off += SZ_BD;
  float* avec  = ws + off; off += SZ_HSD;
  float* avec2 = ws + off; off += SZ_CORE;
  float* ffh   = ws + off; off += SZ_FFH;
  (void)ws_size; (void)n_in; (void)in_sizes; (void)out_size;

  // ---- frontend ----
  {
    const long long n = SZ_CORE;
    conv_pool_kernel<<<(unsigned)((n + 255) / 256), 256, 0, stream>>>(signal, conv_w, conv_b, core);
  }
  posemb_kernel<<<(R_LEN * D_MODEL + 255) / 256, 256, 0, stream>>>(pe);

  const long long sHead = (long long)S_LEN * D_HEAD; // 65472: per-(b,n) stride

  for (int l = 0; l < N_LAYER; ++l) {
    const float* qkv = qkv_w + (long long)l * D_MODEL * 3 * D_MODEL;
    const float* rwl = r_w   + (long long)l * D_MODEL * D_MODEL;
    const float* owl = o_w   + (long long)l * D_MODEL * D_MODEL;
    const float* f1w = ff1_w + (long long)l * D_MODEL * D_INNER;
    const float* f1b = ff1_b + (long long)l * D_INNER;
    const float* f2w = ff2_w + (long long)l * D_INNER * D_MODEL;
    const float* f2b = ff2_b + (long long)l * D_MODEL;

    // heads = core @ qkv_w[l]   (4092 x 1536, K=512)
    gemm(stream, core, qkv, nullptr, heads, nullptr,
         SB, 3 * D_MODEL, D_MODEL, D_MODEL, 3 * D_MODEL, 3 * D_MODEL,
         0, 0, 0, 1, 0, 0);
    // rk = pos_emb @ r_w[l]     (2045 x 512, K=512)
    gemm(stream, pe, rwl, nullptr, rk, nullptr,
         R_LEN, D_MODEL, D_MODEL, D_MODEL, D_MODEL, D_MODEL,
         0, 0, 0, 1, 0, 0);
    // split into per-(b,n) q(+biases)/k/v
    {
      const long long n = SZ_HSD;
      split_qkv_kernel<<<(unsigned)((n + 255) / 256), 256, 0, stream>>>(
          heads, r_w_bias, r_r_bias, qw, qr, kk, vv);
    }

    // attention, chunked over query rows
    for (int c = 0; c < NCHUNK; ++c) {
      const int i0 = c * CHUNK;
      const int rows = (S_LEN - i0 < CHUNK) ? (S_LEN - i0) : CHUNK;
      // AC = (q+rwb) @ k^T : per z, (rows x 1023), K=64
      gemm(stream, qw + (long long)i0 * D_HEAD, kk, nullptr, acb, nullptr,
           rows, S_LEN, D_HEAD, D_HEAD, D_HEAD, LD_AC,
           sHead, sHead, (long long)CHUNK * LD_AC, BN, 0, 2);
      // BD = (q+rrb) @ rk^T : per z, (rows x 2045), K=64; B shared across b (z%8)
      gemm(stream, qr + (long long)i0 * D_HEAD, rk, nullptr, bdb, nullptr,
           rows, R_LEN, D_HEAD, D_HEAD, D_MODEL, LD_BD,
           sHead, (long long)D_HEAD, (long long)CHUNK * LD_BD, BN, N_HEAD, 2);
      // fused rel-shift + softmax -> probabilities in acb
      {
        dim3 g(rows, BN);
        softmax_rel_kernel<<<g, 256, 0, stream>>>(acb, bdb, i0);
      }
      // attn_vec chunk = P @ v : (rows x 64), K=1023
      gemm(stream, acb, vv, nullptr, avec + (long long)i0 * D_HEAD, nullptr,
           rows, D_HEAD, S_LEN, LD_AC, D_HEAD, D_HEAD,
           (long long)CHUNK * LD_AC, sHead, sHead, BN, 0, 0);
    }

    // merge heads to (S*B, 512)
    {
      const long long n = SZ_CORE;
      head_merge_kernel<<<(unsigned)((n + 255) / 256), 256, 0, stream>>>(avec, avec2);
    }
    // core += avec2 @ o_w[l]
    gemm(stream, avec2, owl, core, core, nullptr,
         SB, D_MODEL, D_MODEL, D_MODEL, D_MODEL, D_MODEL,
         0, 0, 0, 1, 0, 4);
    // ffh = relu(core @ ff1_w + ff1_b)
    gemm(stream, core, f1w, nullptr, ffh, f1b,
         SB, D_INNER, D_MODEL, D_MODEL, D_INNER, D_INNER,
         0, 0, 0, 1, 0, 1);
    // core += ffh @ ff2_w + ff2_b
    gemm(stream, ffh, f2w, core, core, f2b,
         SB, D_MODEL, D_INNER, D_INNER, D_MODEL, D_MODEL,
         0, 0, 0, 1, 0, 4);
  }

  fc_kernel<<<BATCH, 16, 0, stream>>>(core, fc_w, fc_b, out);
}
